// Attention_4535485464850
// MI455X (gfx1250) — matrix operand order repaired and checked by reading, not compiled, not run on hardware
//
#include <hip/hip_runtime.h>

typedef __bf16 bf16;
typedef __attribute__((ext_vector_type(8)))  __bf16 bf16x8;
typedef __attribute__((ext_vector_type(16))) __bf16 v16bf;
typedef __attribute__((ext_vector_type(8)))  float  v8f;
typedef __attribute__((ext_vector_type(4)))  unsigned int u32x4;
typedef __attribute__((ext_vector_type(4)))  int i32x4;
typedef __attribute__((ext_vector_type(8)))  int i32x8;

#define EMBED 1024
#define SEQ   2048
#define NB    4
#define NH    16
#define HS    64

#if defined(__has_builtin)
#if __has_builtin(__builtin_amdgcn_tensor_load_to_lds) && __has_builtin(__builtin_amdgcn_s_wait_tensorcnt)
#define USE_TDM 1
#endif
#endif

__device__ __forceinline__ v16bf cat8(bf16x8 lo, bf16x8 hi) {
  return __builtin_shufflevector(lo, hi, 0,1,2,3,4,5,6,7,8,9,10,11,12,13,14,15);
}

__device__ __forceinline__ v8f wmma_bf16(v16bf a, v16bf b, v8f c) {
  // D = A(16x32 bf16) * B(32x16 bf16) + C(16x16 f32)
  return __builtin_amdgcn_wmma_f32_16x16x32_bf16(false, a, false, b, (short)0, c, false, false);
}

#if USE_TDM
// Issue one TDM 2D tile load: tile_d1 rows x tile_d0 elems (bf16), row stride
// stride_d0 elems in memory, packed into LDS with a 16B pad after each 64B row
// (pad_interval=3 -> every 16 DWORDs, pad_amount=3 -> 4 DWORDs) so the LDS
// layout is row-major with stride 40 bf16 == LDSP. D# packing per CDNA5 ISA
// 8.3-8.6. Wave-uniform args; EXEC-independent, tracked by TENSORcnt.
__device__ __forceinline__ void tdm_load_tile(const void* gptr, unsigned lds_byte_addr,
                                              unsigned tensor_d0, unsigned tensor_d1,
                                              unsigned tile_d0, unsigned tile_d1,
                                              unsigned stride_d0) {
  unsigned long long ga = (unsigned long long)(size_t)gptr;
  u32x4 g0;
  g0[0] = 1u;                                              // count=1, no gather
  g0[1] = lds_byte_addr;                                   // D#.lds_addr
  g0[2] = (unsigned)(ga & 0xFFFFFFFFu);                    // global_addr[31:0]
  g0[3] = (unsigned)((ga >> 32) & 0x01FFFFFFu) | (2u << 30); // ga[56:32] | type=2
  i32x8 g1;
  g1[0] = (int)((1u << 16)        // data_size = 2 bytes (bf16)
              | (1u << 20)        // pad_enable
              | (3u << 22)        // pad_interval: 2^(3+1)=16 DWORDs = 64B row
              | (3u << 25));      // pad_amount: 3+1=4 DWORDs = 16B pad
  g1[1] = (int)((tensor_d0 & 0xFFFFu) << 16);              // dim0[15:0] @ [63:48]
  g1[2] = (int)(((tensor_d0 >> 16) & 0xFFFFu) | ((tensor_d1 & 0xFFFFu) << 16));
  g1[3] = (int)(((tensor_d1 >> 16) & 0xFFFFu) | (tile_d0 << 16));
  g1[4] = (int)(tile_d1 & 0xFFFFu);                        // tile_dim1; tile_dim2=0
  g1[5] = (int)stride_d0;                                  // dim0_stride[31:0]
  g1[6] = 0;                                               // stride hi, dim1_stride=0
  g1[7] = 0;
  i32x4 z4 = {0, 0, 0, 0};                                 // groups 2/3: 2D tile
#if __clang_major__ >= 23
  i32x8 z8 = {0, 0, 0, 0, 0, 0, 0, 0};
  __builtin_amdgcn_tensor_load_to_lds(g0, g1, z4, z4, z8, 0);
#else
  __builtin_amdgcn_tensor_load_to_lds(g0, g1, z4, z4, 0);
#endif
}
#endif

// ---------------- fp32 -> bf16 conversion (8 elems / thread) ----------------
__global__ __launch_bounds__(256) void cvt8_kernel(const float* __restrict__ s,
                                                   bf16* __restrict__ d, int n8) {
  int i = blockIdx.x * 256 + threadIdx.x;
  if (i >= n8) return;
  const float4* sp = (const float4*)s + (size_t)i * 2;
  float4 a = sp[0], b = sp[1];
  bf16x8 o;
  o[0]=(bf16)a.x; o[1]=(bf16)a.y; o[2]=(bf16)a.z; o[3]=(bf16)a.w;
  o[4]=(bf16)b.x; o[5]=(bf16)b.y; o[6]=(bf16)b.z; o[7]=(bf16)b.w;
  ((bf16x8*)d)[i] = o;
}

// ------------- weight transpose+convert: W[K][N] f32 -> WT[N][K] bf16 -------
__global__ __launch_bounds__(256) void twt_kernel(const float* __restrict__ w,
                                                  bf16* __restrict__ wt) {
  int idx = blockIdx.x * 256 + threadIdx.x;     // 0 .. 1M-1
  int n = idx >> 10, k = idx & 1023;
  wt[idx] = (bf16)w[(size_t)k * EMBED + n];     // writes coalesced; reads via 192MB L2
}

// ------- 128x128 double-buffered WMMA GEMM: C = A[M,1024] @ W + bias --------
// A: bf16 row-major [M,K=1024], WT: bf16 [N=1024][K] (pre-transposed)
// 8 waves, 32x64 per wave; 1 barrier / K-step. Tile fetch: TDM descriptors
// (wave 0, TENSORcnt) when available, else register-staged global->LDS.
// epi 0: bf16 -> [b,h,t,d]   epi 1: bf16 -> [b,h,d,t]   epi 2: f32 -> [M,N]+bias
#define LDSP 40
__global__ __launch_bounds__(256)
void gemm128x128(const bf16* __restrict__ A, const bf16* __restrict__ WT,
                 const float* __restrict__ bias, void* __restrict__ out, int epi) {
  __shared__ __align__(16) bf16 la[2][128 * LDSP];
  __shared__ __align__(16) bf16 lb[2][128 * LDSP];
  const int K = EMBED;
  const int m0 = blockIdx.x * 128;
  const int n0 = blockIdx.y * 128;
  const int tid  = threadIdx.x;
  const int lane = tid & 31;
  const int wave = tid >> 5;
  const int wm = wave >> 1;               // 0..3 -> 32-row strip
  const int wn = wave & 1;                // 0..1 -> 64-col strip
  const int half = lane >> 4, r = lane & 15;

  v8f zero = {0.f,0.f,0.f,0.f,0.f,0.f,0.f,0.f};
  v8f acc[2][4];
#pragma unroll
  for (int mf = 0; mf < 2; ++mf)
#pragma unroll
    for (int nf = 0; nf < 4; ++nf) acc[mf][nf] = zero;

#if USE_TDM
  const unsigned laddrA[2] = { (unsigned)(size_t)&la[0][0], (unsigned)(size_t)&la[1][0] };
  const unsigned laddrB[2] = { (unsigned)(size_t)&lb[0][0], (unsigned)(size_t)&lb[1][0] };
  if (wave == 0) {                        // prologue: tile 0 -> buf 0 via TDM
    tdm_load_tile(A  + (size_t)m0 * K, laddrA[0], K, NB * SEQ, 32, 128, K);
    tdm_load_tile(WT + (size_t)n0 * K, laddrB[0], K, EMBED,    32, 128, K);
    __builtin_amdgcn_s_wait_tensorcnt(0);
  }
  __syncthreads();
#else
  int ldrow[2], ldc8[2];
#pragma unroll
  for (int it = 0; it < 2; ++it) {
    int idx = tid + it * 256;             // 512 chunks of 8 bf16 per 128x32 tile
    ldrow[it] = idx >> 2;
    ldc8[it]  = (idx & 3) * 8;
  }
  bf16x8 ra[2], rb[2];
#pragma unroll
  for (int it = 0; it < 2; ++it) {
    ra[it] = *(const bf16x8*)&A [(size_t)(m0 + ldrow[it]) * K + ldc8[it]];
    rb[it] = *(const bf16x8*)&WT[(size_t)(n0 + ldrow[it]) * K + ldc8[it]];
  }
#pragma unroll
  for (int it = 0; it < 2; ++it) {
    *(bf16x8*)&la[0][ldrow[it] * LDSP + ldc8[it]] = ra[it];
    *(bf16x8*)&lb[0][ldrow[it] * LDSP + ldc8[it]] = rb[it];
  }
  __syncthreads();
#endif

  const int NT = K / 32;
  for (int i = 0; i < NT; ++i) {
    const int cur = i & 1;
    const bool has_next = (i + 1) < NT;
    if (has_next) {                       // start next tile's fetch now
      const int k0n = (i + 1) * 32;
#if USE_TDM
      if (wave == 0) {
        const int nxt = cur ^ 1;
        tdm_load_tile(A  + (size_t)m0 * K + k0n, laddrA[nxt], K, NB * SEQ, 32, 128, K);
        tdm_load_tile(WT + (size_t)n0 * K + k0n, laddrB[nxt], K, EMBED,    32, 128, K);
      }
#else
#pragma unroll
      for (int it = 0; it < 2; ++it) {
        ra[it] = *(const bf16x8*)&A [(size_t)(m0 + ldrow[it]) * K + k0n + ldc8[it]];
        rb[it] = *(const bf16x8*)&WT[(size_t)(n0 + ldrow[it]) * K + k0n + ldc8[it]];
      }
#endif
    }
    // fragments from current buffer
    v16bf af[2], bfq[4];
#pragma unroll
    for (int mf = 0; mf < 2; ++mf) {
      int arow = wm * 32 + mf * 16 + r;   // A: lane=row, K halves 0-7/16-23 | 8-15/24-31
      af[mf] = cat8(*(const bf16x8*)&la[cur][arow * LDSP + half * 8],
                    *(const bf16x8*)&la[cur][arow * LDSP + 16 + half * 8]);
    }
#pragma unroll
    for (int nf = 0; nf < 4; ++nf) {
      int brow = wn * 64 + nf * 16 + r;   // B: lane=col, K halves 0-7/16-23 | 8-15/24-31
      bfq[nf] = cat8(*(const bf16x8*)&lb[cur][brow * LDSP + half * 8],
                     *(const bf16x8*)&lb[cur][brow * LDSP + 16 + half * 8]);
    }
#pragma unroll
    for (int mf = 0; mf < 2; ++mf)
#pragma unroll
      for (int nf = 0; nf < 4; ++nf)
        acc[mf][nf] = wmma_bf16(af[mf], bfq[nf], acc[mf][nf]);

    if (has_next) {                       // complete next-tile staging
#if USE_TDM
      if (wave == 0) __builtin_amdgcn_s_wait_tensorcnt(0);
#else
      const int nxt = cur ^ 1;
#pragma unroll
      for (int it = 0; it < 2; ++it) {
        *(bf16x8*)&la[nxt][ldrow[it] * LDSP + ldc8[it]] = ra[it];
        *(bf16x8*)&lb[nxt][ldrow[it] * LDSP + ldc8[it]] = rb[it];
      }
#endif
    }
    __syncthreads();
  }

#pragma unroll
  for (int mf = 0; mf < 2; ++mf)
#pragma unroll
    for (int nf = 0; nf < 4; ++nf) {
      int n = n0 + wn * 64 + nf * 16 + r;
      float bv = bias[n];
#pragma unroll
      for (int j = 0; j < 8; ++j) {
        int m = m0 + wm * 32 + mf * 16 + half * 8 + j;  // C: VGPR=row, +8 in upper half
        float v = acc[mf][nf][j] + bv;
        if (epi == 2) {
          ((float*)out)[(size_t)m * EMBED + n] = v;
        } else {
          int b = m >> 11, t = m & (SEQ - 1);
          int h = n >> 6,  d = n & (HS - 1);
          if (epi == 0)
            ((bf16*)out)[(((size_t)(b * NH + h) * SEQ + t) << 6) + d] = (bf16)v;   // [b,h,t,d]
          else
            ((bf16*)out)[(((size_t)(b * NH + h) * HS + d) << 11) + t] = (bf16)v;   // [b,h,d,t]
        }
      }
    }
}

// -------- flash attention: 32 q-rows / wave, 32-key tiles, prefetch ---------
__global__ __launch_bounds__(128)
void flash_attn(const bf16* __restrict__ Q, const bf16* __restrict__ Kb,
                const bf16* __restrict__ Vt, bf16* __restrict__ Ob) {
  __shared__ __align__(16) bf16 pstage[4][32 * 32];  // per-wave P transpose staging
  const int bh = blockIdx.x;              // 0..63
  const int b = bh >> 4, h = bh & 15;
  const int qtile = blockIdx.y;           // 0..15  (128 q rows / block)
  const int lane = threadIdx.x & 31;
  const int wave = threadIdx.x >> 5;
  const int half = lane >> 4, r = lane & 15;
  const int q0 = qtile * 128 + wave * 32; // this wave's 32 q rows

  const size_t headQK = (size_t)bh * SEQ * HS;   // [t][d]
  const size_t headV  = (size_t)bh * HS * SEQ;   // [d][t]

  v16bf qf[2][2];                         // [m-frag][d-step]
#pragma unroll
  for (int mf = 0; mf < 2; ++mf)
#pragma unroll
    for (int f = 0; f < 2; ++f) {
      const bf16* qp = Q + headQK + (size_t)(q0 + mf * 16 + r) * HS + f * 32;
      qf[mf][f] = cat8(*(const bf16x8*)(qp + half * 16),
                       *(const bf16x8*)(qp + half * 16 + 8));
    }

  v8f zero = {0.f,0.f,0.f,0.f,0.f,0.f,0.f,0.f};
  v8f o[2][4];
  float mrow[2][8], lrow[2][8];
#pragma unroll
  for (int mf = 0; mf < 2; ++mf) {
#pragma unroll
    for (int nf = 0; nf < 4; ++nf) o[mf][nf] = zero;
#pragma unroll
    for (int j = 0; j < 8; ++j) { mrow[mf][j] = -1e30f; lrow[mf][j] = 0.f; }
  }

  const float scale = 0.125f;             // 64^-0.5
  const int wend = q0 + 32;               // causal bound (q0 multiple of 32)

  for (int ku = 0; ku < wend; ku += 32) {
    if (ku + 32 < wend) {                 // prefetch next K/V tile (global_prefetch_b8)
      __builtin_prefetch(Kb + headQK + (size_t)(ku + 32 + lane) * HS, 0, 3);
      __builtin_prefetch(Vt + headV + (size_t)lane        * SEQ + ku + 32, 0, 3);
      __builtin_prefetch(Vt + headV + (size_t)(lane + 32) * SEQ + ku + 32, 0, 3);
    }

    // ---- K fragments (shared across both m-frags) ----
    v16bf kb[2][2];                       // [u-frag][d-step]
#pragma unroll
    for (int uf = 0; uf < 2; ++uf) {
      const bf16* kp = Kb + headQK + (size_t)(ku + uf * 16 + r) * HS;
      kb[uf][0] = cat8(*(const bf16x8*)(kp + half * 16),
                       *(const bf16x8*)(kp + half * 16 + 8));
      kb[uf][1] = cat8(*(const bf16x8*)(kp + 32 + half * 16),
                       *(const bf16x8*)(kp + 32 + half * 16 + 8));
    }

    // ---- S = Q K^T : 8 WMMAs ----
    v8f s[2][2];
#pragma unroll
    for (int mf = 0; mf < 2; ++mf)
#pragma unroll
      for (int uf = 0; uf < 2; ++uf) {
        s[mf][uf] = wmma_bf16(qf[mf][0], kb[uf][0], zero);
        s[mf][uf] = wmma_bf16(qf[mf][1], kb[uf][1], s[mf][uf]);
      }

    // ---- scale + causal mask + online softmax (half-wave reductions) ----
    float sv[2][2][8];
#pragma unroll
    for (int mf = 0; mf < 2; ++mf)
#pragma unroll
      for (int uf = 0; uf < 2; ++uf) {
        int u = ku + uf * 16 + r;
#pragma unroll
        for (int j = 0; j < 8; ++j) {
          int q = q0 + mf * 16 + half * 8 + j;
          float v = s[mf][uf][j] * scale;
          sv[mf][uf][j] = (u <= q) ? v : -1e30f;
        }
      }
#pragma unroll
    for (int mf = 0; mf < 2; ++mf)
#pragma unroll
      for (int j = 0; j < 8; ++j) {
        float mx = fmaxf(sv[mf][0][j], sv[mf][1][j]);
#pragma unroll
        for (int off = 1; off < 16; off <<= 1)
          mx = fmaxf(mx, __shfl_xor(mx, off, 32));
        float mn = fmaxf(mrow[mf][j], mx);
        float alpha = __expf(mrow[mf][j] - mn);
        mrow[mf][j] = mn;
        float p0 = __expf(sv[mf][0][j] - mn);
        float p1 = __expf(sv[mf][1][j] - mn);
        sv[mf][0][j] = p0; sv[mf][1][j] = p1;
        float psum = p0 + p1;
#pragma unroll
        for (int off = 1; off < 16; off <<= 1)
          psum += __shfl_xor(psum, off, 32);
        lrow[mf][j] = lrow[mf][j] * alpha + psum;
#pragma unroll
        for (int nf = 0; nf < 4; ++nf) o[mf][nf][j] *= alpha;
      }

    // ---- transpose P (C layout -> A layout) via per-wave LDS ----
    bf16* ps = &pstage[wave][0];
#pragma unroll
    for (int mf = 0; mf < 2; ++mf)
#pragma unroll
      for (int uf = 0; uf < 2; ++uf)
#pragma unroll
        for (int j = 0; j < 8; ++j)
          ps[(mf * 16 + half * 8 + j) * 32 + uf * 16 + r] = (bf16)sv[mf][uf][j];
    asm volatile("s_wait_dscnt 0" ::: "memory");   // same-wave LDS RAW fence
    v16bf pa[2];
#pragma unroll
    for (int mf = 0; mf < 2; ++mf) {
      const bf16* pr = ps + (mf * 16 + r) * 32;
      pa[mf] = cat8(*(const bf16x8*)(pr + half * 16),
                    *(const bf16x8*)(pr + half * 16 + 8));
    }

    // ---- O += P V : V fragments reused across both m-frags, 8 WMMAs ----
#pragma unroll
    for (int nf = 0; nf < 4; ++nf) {
      const bf16* vp = Vt + headV + (size_t)(nf * 16 + r) * SEQ + ku;
      v16bf vb = cat8(*(const bf16x8*)(vp + half * 16),
                      *(const bf16x8*)(vp + half * 16 + 8));
#pragma unroll
      for (int mf = 0; mf < 2; ++mf)
        o[mf][nf] = wmma_bf16(pa[mf], vb, o[mf][nf]);
    }
  }

  // ---- finalize: O/l -> attn buffer [b, t, h*64+d] bf16 ----
#pragma unroll
  for (int mf = 0; mf < 2; ++mf)
#pragma unroll
    for (int nf = 0; nf < 4; ++nf) {
      int d = nf * 16 + r;
#pragma unroll
      for (int j = 0; j < 8; ++j) {
        int q = q0 + mf * 16 + half * 8 + j;
        float v = o[mf][nf][j] / lrow[mf][j];
        Ob[((size_t)b * SEQ + q) * EMBED + h * HS + d] = (bf16)v;
      }
    }
}

extern "C" void kernel_launch(void* const* d_in, const int* in_sizes, int n_in,
                              void* d_out, int out_size, void* d_ws, size_t ws_size,
                              hipStream_t stream) {
  (void)in_sizes; (void)n_in; (void)out_size; (void)ws_size;
  const float* x  = (const float*)d_in[0];
  const float* wq = (const float*)d_in[1];
  const float* bq = (const float*)d_in[2];
  const float* wk = (const float*)d_in[3];
  const float* bk = (const float*)d_in[4];
  const float* wv = (const float*)d_in[5];
  const float* bv = (const float*)d_in[6];
  const float* wo = (const float*)d_in[7];
  const float* bo = (const float*)d_in[8];

  char* ws = (char*)d_ws;                       // 88 MB layout
  bf16* xb    = (bf16*)ws;                      // 16 MB  x bf16 [8192,1024]
  bf16* wqT   = (bf16*)(ws + (16u << 20));      //  2 MB each, [N][K]
  bf16* wkT   = wqT + (1u << 20);
  bf16* wvT   = wkT + (1u << 20);
  bf16* woT   = wvT + (1u << 20);
  bf16* Qb    = (bf16*)(ws + (24u << 20));      // 16 MB [b,h,t,d]
  bf16* Kb    = Qb + (8u << 20);                // 16 MB [b,h,t,d]
  bf16* Vt    = Kb + (8u << 20);                // 16 MB [b,h,d,t]
  bf16* attnb = Vt + (8u << 20);                // 16 MB [b,t,c]

  const int M = NB * SEQ;                       // 8192
  cvt8_kernel<<<(M * EMBED / 8 + 255) / 256, 256, 0, stream>>>(x, xb, M * EMBED / 8);
  twt_kernel<<<4096, 256, 0, stream>>>(wq, wqT);
  twt_kernel<<<4096, 256, 0, stream>>>(wk, wkT);
  twt_kernel<<<4096, 256, 0, stream>>>(wv, wvT);
  twt_kernel<<<4096, 256, 0, stream>>>(wo, woT);

  dim3 gg(M / 128, EMBED / 128);
  gemm128x128<<<gg, 256, 0, stream>>>(xb, wqT, bq, Qb, 0);
  gemm128x128<<<gg, 256, 0, stream>>>(xb, wkT, bk, Kb, 0);
  gemm128x128<<<gg, 256, 0, stream>>>(xb, wvT, bv, Vt, 1);

  flash_attn<<<dim3(NB * NH, SEQ / 128), 128, 0, stream>>>(Qb, Kb, Vt, attnb);

  gemm128x128<<<gg, 256, 0, stream>>>(attnb, woT, bo, d_out, 2);
}
